// SparseTechnicalNetwork_62981400429240
// MI455X (gfx1250) — matrix operand order repaired and checked by reading, not compiled, not run on hardware
//
#include <hip/hip_runtime.h>
#include <hip/hip_bf16.h>
#include <cstdint>
#include <cstddef>

// ---------------------------------------------------------------------------
// MI455X (gfx1250) implementation of SparseTechnicalNetwork forward.
//
//  * LSTM recurrence: per-step gates = [h | x] @ [w_hh ; w_ih].T via
//    v_wmma_f32_16x16x32_f16. Packed B-fragments become register-resident
//    across the T loop (compiler hoists, extended VGPR file).
//  * Layer-1 per-step input tile streamed with global_load_async_to_lds_b128
//    (double-buffered, ASYNCcnt) to overlap with WMMA compute.
//  * gather-einsum collapsed to dense GEMM: z = base @ Wz.T with
//    Wz[n,j] = sum_{k: conn_idx%32==j} conn_w[n,k].
//  * One generic WMMA GEMM kernel with fused epilogues drives the tail.
// ---------------------------------------------------------------------------

typedef __attribute__((ext_vector_type(16))) _Float16 v16h;
typedef __attribute__((ext_vector_type(8)))  float    v8f;
typedef __attribute__((ext_vector_type(4)))  unsigned int u32x4;

union Frag { u32x4 u[2]; v16h h; };

__device__ inline float sigf(float x) { return 1.0f / (1.0f + __expf(-x)); }
__device__ inline float tanhfast(float x) {
    x = fminf(fmaxf(x, -15.0f), 15.0f);
    float e = __expf(2.0f * x);
    return (e - 1.0f) / (e + 1.0f);
}
__device__ inline void zero8(v8f& v) {
#pragma unroll
    for (int i = 0; i < 8; ++i) v[i] = 0.0f;
}

// Async global->LDS 16B copy (per lane), tracked by ASYNCcnt.
__device__ inline void async_load_b128(unsigned lds_off, unsigned gvoff,
                                       const void* sbase) {
    asm volatile("global_load_async_to_lds_b128 %0, %1, %2"
                 :: "v"(lds_off), "v"(gvoff),
                    "s"((unsigned long long)(size_t)sbase)
                 : "memory");
}
__device__ inline void wait_asynccnt0() {
    asm volatile("s_wait_asynccnt 0x0" ::: "memory");
}

static constexpr int BATCH = 1024;
static constexpr int TT    = 60;
static constexpr int HH    = 128;

// ---------------------------------------------------------------------------
// Weight packing into WMMA B-fragment layout.
// B matrix (K x N). Fragment (ktile, ntile): lane l holds column
// n = nt*16 + (l&15), K values k = kt*32 + 16*(l>>4) + v, v = 0..15,
// stored contiguously: dst[(((nt*KT)+kt)*32 + l)*16 + v].
// ---------------------------------------------------------------------------
__global__ void pack_b_generic(const float* __restrict__ W, int N, int K,
                               int KT, int total, _Float16* __restrict__ dst) {
    for (int idx = blockIdx.x * blockDim.x + threadIdx.x; idx < total;
         idx += gridDim.x * blockDim.x) {
        int v    = idx & 15;
        int lane = (idx >> 4) & 31;
        int rest = idx >> 9;
        int kt   = rest % KT;
        int nt   = rest / KT;
        int n = nt * 16 + (lane & 15);
        int k = kt * 32 + 16 * (lane >> 4) + v;
        float val = (n < N && k < K) ? W[(size_t)n * K + k] : 0.0f;
        dst[idx] = (_Float16)val;
    }
}

// LSTM combined B: rows 0..127 = w_hh[n][k]; layer0 rows 128..132 = w_ih
// (F=5, rest zero-padded to K=160); layer1 rows 128..255 = w_ih (K=256).
template <bool LAYER1>
__global__ void pack_b_lstm(const float* __restrict__ w_hh,
                            const float* __restrict__ w_ih,
                            int KT, int total, _Float16* __restrict__ dst) {
    for (int idx = blockIdx.x * blockDim.x + threadIdx.x; idx < total;
         idx += gridDim.x * blockDim.x) {
        int v    = idx & 15;
        int lane = (idx >> 4) & 31;
        int rest = idx >> 9;
        int kt   = rest % KT;
        int nt   = rest / KT;
        int n = nt * 16 + (lane & 15);          // gate index 0..511
        int k = kt * 32 + 16 * (lane >> 4) + v; // contraction index
        float val = 0.0f;
        if (k < 128) val = w_hh[(size_t)n * 128 + k];
        else if (!LAYER1) { if (k < 133) val = w_ih[(size_t)n * 5 + (k - 128)]; }
        else              { if (k < 256) val = w_ih[(size_t)n * 128 + (k - 128)]; }
        dst[idx] = (_Float16)val;
    }
}

__global__ void vec_add(const float* __restrict__ a, const float* __restrict__ b,
                        int n, float* __restrict__ d) {
    int i = blockIdx.x * blockDim.x + threadIdx.x;
    if (i < n) d[i] = a[i] + b[i];
}

// Wz[n][j] = sum over k of conn_w[n,k] where conn_idx[n,k] % 32 == j.
__global__ void build_wz(const int* __restrict__ conn_idx,
                         const float* __restrict__ conn_w,
                         float* __restrict__ Wz) {
    int n = blockIdx.x * blockDim.x + threadIdx.x;
    if (n >= 2500) return;
    float acc[32];
#pragma unroll
    for (int j = 0; j < 32; ++j) acc[j] = 0.0f;
    for (int k = 0; k < 50; ++k) {
        int j = conn_idx[n * 50 + k] & 31;
        acc[j] += conn_w[n * 50 + k];
    }
#pragma unroll
    for (int j = 0; j < 32; ++j) Wz[(size_t)n * 32 + j] = acc[j];
}

// Concatenate 5 head weight matrices into Wh (16 x 32) + bias (16), row 15 = 0.
__global__ void build_head(const float* wt, const float* wp, const float* wk,
                           const float* wv, const float* wc,
                           const float* bt, const float* bp, const float* bk,
                           const float* bv, const float* bc,
                           float* __restrict__ Wh, float* __restrict__ bh) {
    int idx = threadIdx.x;
    if (idx < 512) {
        int r = idx >> 5, c = idx & 31;
        float v = 0.0f;
        if      (r < 3)   v = wt[r * 32 + c];
        else if (r < 9)   v = wp[(r - 3) * 32 + c];
        else if (r < 13)  v = wk[(r - 9) * 32 + c];
        else if (r == 13) v = wv[c];
        else if (r == 14) v = wc[c];
        Wh[idx] = v;
    }
    if (idx < 16) {
        float v = 0.0f;
        if      (idx < 3)   v = bt[idx];
        else if (idx < 9)   v = bp[idx - 3];
        else if (idx < 13)  v = bk[idx - 9];
        else if (idx == 13) v = bv[0];
        else if (idx == 14) v = bc[0];
        bh[idx] = v;
    }
}

// ---------------------------------------------------------------------------
// LSTM layer: block = 16 batch rows x full 512 gate cols, 8 waves.
// A = [h(128) | x-part]: h fp16 in Hlds, x in Xlds (layer1: async double-
// buffered). B fragments register-resident. c in LDS f32.
// ---------------------------------------------------------------------------
template <int KA, bool LAYER1>
__global__ __launch_bounds__(256)
void lstm_layer_kernel(const float* __restrict__ x,          // (B,T,5) layer0
                       const _Float16* __restrict__ seq_in,  // (B,T,128) layer1
                       const _Float16* __restrict__ Bp,      // packed weights
                       const float* __restrict__ bias,       // b_ih + b_hh (512)
                       _Float16* __restrict__ seq_out,       // (B,T,128) layer0
                       _Float16* __restrict__ h_final) {     // (B,128) layer1
    constexpr int KT  = KA / 32;       // total k-tiles (5 or 8)
    constexpr int KHT = 4;             // k-tiles covering the h part (128)
    constexpr int XW  = KA - 128;      // x-part width in halfs (32 or 128)
    constexpr int NXB = LAYER1 ? 2 : 1;

    __shared__ __align__(16) _Float16 Hlds[16 * 128];
    __shared__ __align__(16) _Float16 Xlds[NXB * 16 * XW];
    __shared__ float gates[16 * 512];
    __shared__ float clds[16 * 128];

    const int tid  = threadIdx.x;
    const int lane = tid & 31;
    const int wave = tid >> 5;
    const int b0   = blockIdx.x * 16;

    const int l16   = lane & 15;
    const int half  = lane >> 4;
    const int abase = half * 16;
    const int ntb   = wave * 4;

    // 16B-chunk indices for vectorized tile IO (one b128 per thread).
    const int pr = tid >> 4;           // row 0..15
    const int pc = (tid & 15) * 8;     // half-column 0,8,...,120

    // Per-thread fixed gate column for the activation phase.
    const int aj = tid & 127;
    const int r0 = tid >> 7;
    const float bI = bias[aj];
    const float bF = bias[128 + aj];
    const float bG = bias[256 + aj];
    const float bO = bias[384 + aj];

    for (int i = tid; i < 16 * 128; i += 256) Hlds[i] = (_Float16)0.0f;
    for (int i = tid; i < 16 * 128; i += 256) clds[i] = 0.0f;
    if (!LAYER1)
        for (int i = tid; i < 16 * XW; i += 256) Xlds[i] = (_Float16)0.0f;

    if (LAYER1) {
        // Prologue: async-prefetch tile t=0 into buffer 0.
        unsigned lds_off = (unsigned)(size_t)&Xlds[pr * XW + pc];
        unsigned gv = (unsigned)((((unsigned)(b0 + pr) * TT + 0) * HH + pc) * 2);
        async_load_b128(lds_off, gv, seq_in);
    }
    __syncthreads();

    for (int t = 0; t < TT; ++t) {
        // ---- phase A: make this step's A matrix available -------------
        if (LAYER1) {
            wait_asynccnt0();          // tile t is now in Xlds[t&1]
            if (t + 1 < TT) {          // overlap tile t+1 with this step
                unsigned lds_off = (unsigned)(size_t)
                    &Xlds[((t + 1) & 1) * 16 * XW + pr * XW + pc];
                unsigned gv = (unsigned)
                    ((((unsigned)(b0 + pr) * TT + (t + 1)) * HH + pc) * 2);
                async_load_b128(lds_off, gv, seq_in);
            }
        } else {
            if (t > 0) {               // vectorized seq_out store of h(t-1)
                u32x4 hv = *(const u32x4*)&Hlds[pr * 128 + pc];
                *(u32x4*)&seq_out[((size_t)(b0 + pr) * TT + (t - 1)) * HH + pc] = hv;
            }
            if (tid < 80) {
                int r = tid / 5, f = tid % 5;
                Xlds[r * XW + f] =
                    (_Float16)x[((size_t)(b0 + r) * TT + t) * 5 + f];
            }
        }
        __syncthreads();

        // ---- gates = A @ B via WMMA; each wave owns 4 n-tiles ----------
        const _Float16* Xcur = &Xlds[(LAYER1 ? (t & 1) : 0) * 16 * XW];

        Frag afr[KT];
#pragma unroll
        for (int kt = 0; kt < KT; ++kt) {
            const _Float16* src = (kt < KHT)
                ? &Hlds[l16 * 128 + kt * 32 + abase]
                : &Xcur[l16 * XW + (kt - KHT) * 32 + abase];
            afr[kt].u[0] = *(const u32x4*)(src);
            afr[kt].u[1] = *(const u32x4*)(src + 8);
        }

        v8f acc[4];
#pragma unroll
        for (int j = 0; j < 4; ++j) zero8(acc[j]);

#pragma unroll
        for (int kt = 0; kt < KT; ++kt) {
#pragma unroll
            for (int j = 0; j < 4; ++j) {
                Frag bf;
                const _Float16* bp =
                    Bp + (((size_t)(ntb + j) * KT + kt) * 32 + lane) * 16;
                bf.u[0] = ((const u32x4*)bp)[0];
                bf.u[1] = ((const u32x4*)bp)[1];
                acc[j] = __builtin_amdgcn_wmma_f32_16x16x32_f16(
                    false, afr[kt].h, false, bf.h, (short)0, acc[j], false, false);
            }
        }
#pragma unroll
        for (int j = 0; j < 4; ++j) {
            int n = (ntb + j) * 16 + l16;
#pragma unroll
            for (int i = 0; i < 8; ++i) {
                int m = half * 8 + i;
                gates[m * 512 + n] = acc[j][i];
            }
        }
        __syncthreads();

        // ---- LSTM cell activations; h back into Hlds (fp16) ------------
#pragma unroll
        for (int q = 0; q < 8; ++q) {
            int r = r0 + 2 * q;
            float ii = gates[r * 512 + aj]       + bI;
            float ff = gates[r * 512 + 128 + aj] + bF;
            float gg = gates[r * 512 + 256 + aj] + bG;
            float oo = gates[r * 512 + 384 + aj] + bO;
            float c  = sigf(ff) * clds[r * 128 + aj] + sigf(ii) * tanhfast(gg);
            float h  = sigf(oo) * tanhfast(c);
            clds[r * 128 + aj] = c;
            Hlds[r * 128 + aj] = (_Float16)h;
        }
        __syncthreads();
    }

    // Epilogue: final h tile (vectorized).
    u32x4 hv = *(const u32x4*)&Hlds[pr * 128 + pc];
    if (!LAYER1)
        *(u32x4*)&seq_out[((size_t)(b0 + pr) * TT + (TT - 1)) * HH + pc] = hv;
    else
        *(u32x4*)&h_final[(size_t)(b0 + pr) * HH + pc] = hv;
}

// ---------------------------------------------------------------------------
// Generic WMMA GEMM: out = epi(A(f16, M x lda) @ Bpacked + bias).
// Block: 256 threads = 8 waves, covers 128 rows x 64 cols.
// grid = (M/128, ntiles/4). K = KT*32 == lda (zero-padded).
// ---------------------------------------------------------------------------
enum { EPI_NONE = 0, EPI_RELU = 1, EPI_TANH = 2, EPI_GROUP = 3 };

template <int EPI, bool OUTF32>
__global__ __launch_bounds__(256)
void gemm_wmma_kernel(const _Float16* __restrict__ A, int lda,
                      const _Float16* __restrict__ Bp, int KT,
                      const float* __restrict__ bias, int Nreal, int ldo,
                      _Float16* __restrict__ outH, float* __restrict__ outF,
                      const float* __restrict__ sens,
                      const float* __restrict__ thr) {
    const int tid  = threadIdx.x;
    const int lane = tid & 31;
    const int wave = tid >> 5;
    const int l16   = lane & 15;
    const int half  = lane >> 4;
    const int abase = half * 16;
    const int mrow  = blockIdx.x * 128 + wave * 16;
    const int ntb   = blockIdx.y * 4;

    v8f acc[4];
#pragma unroll
    for (int j = 0; j < 4; ++j) zero8(acc[j]);

#pragma unroll 2
    for (int kt = 0; kt < KT; ++kt) {
        Frag a;
        const _Float16* ap = A + (size_t)(mrow + l16) * lda + kt * 32 + abase;
        a.u[0] = *(const u32x4*)(ap);
        a.u[1] = *(const u32x4*)(ap + 8);
#pragma unroll
        for (int j = 0; j < 4; ++j) {
            Frag bf;
            const _Float16* bp =
                Bp + (((size_t)(ntb + j) * KT + kt) * 32 + lane) * 16;
            bf.u[0] = ((const u32x4*)bp)[0];
            bf.u[1] = ((const u32x4*)bp)[1];
            acc[j] = __builtin_amdgcn_wmma_f32_16x16x32_f16(
                false, a.h, false, bf.h, (short)0, acc[j], false, false);
        }
    }

#pragma unroll
    for (int j = 0; j < 4; ++j) {
        int n = (ntb + j) * 16 + l16;
        float bi = 0.0f, sv = 1.0f, tv = 0.0f;
        if (EPI == EPI_GROUP) {
            if (n < Nreal) { sv = sens[n]; tv = thr[n]; }
        } else if (bias != nullptr && n < Nreal) {
            bi = bias[n];
        }
#pragma unroll
        for (int i = 0; i < 8; ++i) {
            int m = mrow + half * 8 + i;
            float v = acc[j][i];
            if (EPI == EPI_GROUP) {
                float z = v * sv;
                if (n >= Nreal)      v = 0.0f;
                else if (n < 800)    v = sigf(z - tv);
                else if (n < 1500)   v = tanhfast(z);
                else if (n < 2100)   v = fmaxf(z - tv, 0.0f);
                else                 v = sigf(z);
            } else {
                v += bi;
                if (EPI == EPI_RELU)      v = fmaxf(v, 0.0f);
                else if (EPI == EPI_TANH) v = tanhfast(v);
                if (n >= Nreal) v = 0.0f;   // keep pad columns zero
            }
            if (OUTF32) outF[(size_t)m * ldo + n] = v;
            else        outH[(size_t)m * ldo + n] = (_Float16)v;
        }
    }
}

// ---------------------------------------------------------------------------
// Group means of `a` -> out columns 16..19.
// ---------------------------------------------------------------------------
__global__ __launch_bounds__(128)
void means_kernel(const _Float16* __restrict__ a, int lda,
                  float* __restrict__ out) {
    __shared__ float red[128];
    const int b = blockIdx.x;
    const int tid = threadIdx.x;
    const int lo[4] = {0, 800, 1500, 2100};
    const int hi[4] = {800, 1500, 2100, 2500};
    for (int g = 0; g < 4; ++g) {
        float s = 0.0f;
        for (int n = lo[g] + tid; n < hi[g]; n += 128)
            s += (float)a[(size_t)b * lda + n];
        red[tid] = s;
        __syncthreads();
        for (int off = 64; off > 0; off >>= 1) {
            if (tid < off) red[tid] += red[tid + off];
            __syncthreads();
        }
        if (tid == 0) out[b * 20 + 16 + g] = red[0] / (float)(hi[g] - lo[g]);
        __syncthreads();
    }
}

__global__ void assemble_kernel(const float* __restrict__ heads, int ldh,
                                float* __restrict__ out) {
    int idx = blockIdx.x * blockDim.x + threadIdx.x;
    if (idx >= BATCH * 16) return;
    int b = idx >> 4, c = idx & 15;
    float v = (c < 15) ? heads[(size_t)b * ldh + c]
                       : sigf(heads[(size_t)b * ldh + 14]);
    out[b * 20 + c] = v;
}

// ---------------------------------------------------------------------------
// Host-side orchestration.
// ---------------------------------------------------------------------------
extern "C" void kernel_launch(void* const* d_in, const int* in_sizes, int n_in,
                              void* d_out, int out_size, void* d_ws, size_t ws_size,
                              hipStream_t stream) {
    (void)in_sizes; (void)n_in; (void)out_size; (void)ws_size;

    const float* x        = (const float*)d_in[0];
    const int*   conn_idx = (const int*)  d_in[1];
    const float* w_ih0    = (const float*)d_in[2];
    const float* w_hh0    = (const float*)d_in[3];
    const float* b_ih0    = (const float*)d_in[4];
    const float* b_hh0    = (const float*)d_in[5];
    const float* w_ih1    = (const float*)d_in[6];
    const float* w_hh1    = (const float*)d_in[7];
    const float* b_ih1    = (const float*)d_in[8];
    const float* b_hh1    = (const float*)d_in[9];
    const float* pw1      = (const float*)d_in[10];
    const float* pb1      = (const float*)d_in[11];
    const float* pw2      = (const float*)d_in[12];
    const float* pb2      = (const float*)d_in[13];
    const float* conn_w   = (const float*)d_in[14];
    const float* sens     = (const float*)d_in[15];
    const float* thr      = (const float*)d_in[16];
    const float* iw1      = (const float*)d_in[17];
    const float* ib1      = (const float*)d_in[18];
    const float* iw2      = (const float*)d_in[19];
    const float* ib2      = (const float*)d_in[20];
    const float* iw3      = (const float*)d_in[21];
    const float* ib3      = (const float*)d_in[22];
    const float* hw_trend = (const float*)d_in[23];
    const float* hb_trend = (const float*)d_in[24];
    const float* hw_pat   = (const float*)d_in[25];
    const float* hb_pat   = (const float*)d_in[26];
    const float* hw_key   = (const float*)d_in[27];
    const float* hb_key   = (const float*)d_in[28];
    const float* hw_vol   = (const float*)d_in[29];
    const float* hb_vol   = (const float*)d_in[30];
    const float* hw_conf  = (const float*)d_in[31];
    const float* hb_conf  = (const float*)d_in[32];
    float* out = (float*)d_out;

    // ---- workspace layout ----
    char* ws = (char*)d_ws;
    size_t off = 0;
    auto alloc = [&](size_t bytes) -> char* {
        char* p = ws + off;
        off += (bytes + 255) & ~(size_t)255;
        return p;
    };
    _Float16* bp_l0  = (_Float16*)alloc((size_t)32 * 5  * 512 * 2); // lstm0 B
    _Float16* bp_l1  = (_Float16*)alloc((size_t)32 * 8  * 512 * 2); // lstm1 B
    float*    bias0  = (float*)   alloc(512 * 4);
    float*    bias1  = (float*)   alloc(512 * 4);
    _Float16* seq0   = (_Float16*)alloc((size_t)BATCH * TT * HH * 2);
    _Float16* h2     = (_Float16*)alloc((size_t)BATCH * HH * 2);
    _Float16* bp_pw1 = (_Float16*)alloc((size_t)4 * 4 * 512 * 2);
    _Float16* bp_pw2 = (_Float16*)alloc((size_t)4 * 2 * 512 * 2);
    _Float16* g1buf  = (_Float16*)alloc((size_t)BATCH * 64 * 2);
    _Float16* basebf = (_Float16*)alloc((size_t)BATCH * 64 * 2);
    float*    wz     = (float*)   alloc((size_t)2500 * 32 * 4);
    _Float16* bp_wz  = (_Float16*)alloc((size_t)160 * 2 * 512 * 2);
    _Float16* abuf   = (_Float16*)alloc((size_t)BATCH * 2560 * 2);
    _Float16* bp_iw1 = (_Float16*)alloc((size_t)16 * 80 * 512 * 2);
    _Float16* t1buf  = (_Float16*)alloc((size_t)BATCH * 256 * 2);
    _Float16* bp_iw2 = (_Float16*)alloc((size_t)4 * 8 * 512 * 2);
    _Float16* t2buf  = (_Float16*)alloc((size_t)BATCH * 64 * 2);
    _Float16* bp_iw3 = (_Float16*)alloc((size_t)4 * 2 * 512 * 2);
    _Float16* integ  = (_Float16*)alloc((size_t)BATCH * 64 * 2);
    float*    whead  = (float*)   alloc(16 * 32 * 4);
    float*    bhead  = (float*)   alloc(16 * 4);
    _Float16* bp_wh  = (_Float16*)alloc((size_t)4 * 2 * 512 * 2);
    float*    heads  = (float*)   alloc((size_t)BATCH * 64 * 4);

    auto pgrid = [](int total) { return dim3((total + 255) / 256); };

    // ---- pack weights ----
    { int tot = 32 * 5 * 512;
      pack_b_lstm<false><<<pgrid(tot), 256, 0, stream>>>(w_hh0, w_ih0, 5, tot, bp_l0); }
    { int tot = 32 * 8 * 512;
      pack_b_lstm<true ><<<pgrid(tot), 256, 0, stream>>>(w_hh1, w_ih1, 8, tot, bp_l1); }
    vec_add<<<pgrid(512), 256, 0, stream>>>(b_ih0, b_hh0, 512, bias0);
    vec_add<<<pgrid(512), 256, 0, stream>>>(b_ih1, b_hh1, 512, bias1);
    { int tot = 4 * 4 * 512;
      pack_b_generic<<<pgrid(tot), 256, 0, stream>>>(pw1, 64, 128, 4, tot, bp_pw1); }
    { int tot = 4 * 2 * 512;
      pack_b_generic<<<pgrid(tot), 256, 0, stream>>>(pw2, 32, 64, 2, tot, bp_pw2); }
    build_wz<<<pgrid(2500), 256, 0, stream>>>(conn_idx, conn_w, wz);
    { int tot = 160 * 2 * 512;
      pack_b_generic<<<pgrid(tot), 256, 0, stream>>>(wz, 2500, 32, 2, tot, bp_wz); }
    { int tot = 16 * 80 * 512;
      pack_b_generic<<<pgrid(tot), 256, 0, stream>>>(iw1, 256, 2500, 80, tot, bp_iw1); }
    { int tot = 4 * 8 * 512;
      pack_b_generic<<<pgrid(tot), 256, 0, stream>>>(iw2, 64, 256, 8, tot, bp_iw2); }
    { int tot = 4 * 2 * 512;
      pack_b_generic<<<pgrid(tot), 256, 0, stream>>>(iw3, 32, 64, 2, tot, bp_iw3); }
    build_head<<<1, 512, 0, stream>>>(hw_trend, hw_pat, hw_key, hw_vol, hw_conf,
                                      hb_trend, hb_pat, hb_key, hb_vol, hb_conf,
                                      whead, bhead);
    { int tot = 4 * 2 * 512;
      pack_b_generic<<<pgrid(tot), 256, 0, stream>>>(whead, 16, 32, 2, tot, bp_wh); }

    // ---- LSTM layers (sequential scan, WMMA gate GEMMs) ----
    lstm_layer_kernel<160, false><<<64, 256, 0, stream>>>(
        x, nullptr, bp_l0, bias0, seq0, nullptr);
    lstm_layer_kernel<256, true ><<<64, 256, 0, stream>>>(
        nullptr, seq0, bp_l1, bias1, nullptr, h2);

    // ---- feed-forward tail (all WMMA) ----
    // g1 = relu(h2 @ pw1.T + pb1)            M=1024 K=128 N=64
    gemm_wmma_kernel<EPI_RELU, false><<<dim3(8, 1), 256, 0, stream>>>(
        h2, 128, bp_pw1, 4, pb1, 64, 64, g1buf, nullptr, nullptr, nullptr);
    // base = tanh(g1 @ pw2.T + pb2)          K=64 N=32(pad64)
    gemm_wmma_kernel<EPI_TANH, false><<<dim3(8, 1), 256, 0, stream>>>(
        g1buf, 64, bp_pw2, 2, pb2, 32, 64, basebf, nullptr, nullptr, nullptr);
    // a = group_act((base @ Wz.T) * sens)    K=32(pad64) N=2500(pad2560)
    gemm_wmma_kernel<EPI_GROUP, false><<<dim3(8, 40), 256, 0, stream>>>(
        basebf, 64, bp_wz, 2, nullptr, 2500, 2560, abuf, nullptr, sens, thr);
    // t1 = relu(a @ iw1.T + ib1)             K=2500(pad2560) N=256
    gemm_wmma_kernel<EPI_RELU, false><<<dim3(8, 4), 256, 0, stream>>>(
        abuf, 2560, bp_iw1, 80, ib1, 256, 256, t1buf, nullptr, nullptr, nullptr);
    // t2 = relu(t1 @ iw2.T + ib2)            K=256 N=64
    gemm_wmma_kernel<EPI_RELU, false><<<dim3(8, 1), 256, 0, stream>>>(
        t1buf, 256, bp_iw2, 8, ib2, 64, 64, t2buf, nullptr, nullptr, nullptr);
    // integ = tanh(t2 @ iw3.T + ib3)         K=64 N=32(pad64)
    gemm_wmma_kernel<EPI_TANH, false><<<dim3(8, 1), 256, 0, stream>>>(
        t2buf, 64, bp_iw3, 2, ib3, 32, 64, integ, nullptr, nullptr, nullptr);
    // heads = integ @ Wh.T + bh (f32 out)    K=32(pad64) N=16(pad64)
    gemm_wmma_kernel<EPI_NONE, true><<<dim3(8, 1), 256, 0, stream>>>(
        integ, 64, bp_wh, 2, bhead, 16, 64, nullptr, heads, nullptr, nullptr);

    // ---- final assembly ----
    means_kernel<<<BATCH, 128, 0, stream>>>(abuf, 2560, out);
    assemble_kernel<<<pgrid(BATCH * 16), 256, 0, stream>>>(heads, 64, out);
}
